// MetaGIN_43490838839337
// MI455X (gfx1250) — matrix operand order repaired and checked by reading, not compiled, not run on hardware
//
#include <hip/hip_runtime.h>
#include <hip/hip_bf16.h>

// ---------------------------------------------------------------------------
// MetaGIN (5-layer GIN conv) for MI455X / gfx1250, wave32 + WMMA f16.
//
// Pipeline (all on `stream`):
//   once:  scatter-add edge_attr -> aggEA[100K,16], deg[]; build CSR; x->f16
//   per layer:
//     cvt W1,W2 -> f16
//     agg:   aggh[n, 0:128]  = sum_{src in CSR[n]} hf[src] + hf[n]    (f16)
//            aggh[n,128:256] = aggEA[n] @ EWl^T + (deg+1)*EBl          (f16)
//     gemm1: hbig = aggh @ W1l^T + B1l            (WMMA f16->f32, K=256)
//     bn:    column mean/var (biased), scale/shift, fused BN+ReLU -> hact f16
//     gemm2: out  = hact @ W2l^T + B2l            (WMMA f16->f32, K=256)
//            layers 0-3: relu -> hf (f16); layer 4: f32 -> d_out
// ---------------------------------------------------------------------------

typedef __attribute__((ext_vector_type(16))) _Float16 v16h;
typedef __attribute__((ext_vector_type(8)))  float    v8f;
typedef __attribute__((ext_vector_type(4)))  _Float16 v4h;
typedef __attribute__((ext_vector_type(4)))  unsigned int v4u;

union FragH { v16h h; v4u u[2]; };

#define N_NODES 100000
#define N_EDGES 1600000

// ---------------- utility kernels ----------------
__global__ void zero_f32_kernel(float* p, int n) {
    int i = blockIdx.x * blockDim.x + threadIdx.x;
    if (i < n) p[i] = 0.0f;
}
__global__ void zero_i32_kernel(int* p, int n) {
    int i = blockIdx.x * blockDim.x + threadIdx.x;
    if (i < n) p[i] = 0;
}
__global__ void cvt_f32_to_f16_kernel(const float* __restrict__ in,
                                      _Float16* __restrict__ out, int n) {
    int i = blockIdx.x * blockDim.x + threadIdx.x;
    if (i < n) out[i] = (_Float16)in[i];
}

// ---------------- graph preprocessing ----------------
// one thread per (edge, k): scatter-add edge_attr into aggEA, count degree.
__global__ void edge_accum_kernel(const int* __restrict__ dst,
                                  const float* __restrict__ ea,
                                  float* __restrict__ aggEA,
                                  int* __restrict__ deg, int nEdges) {
    int t = blockIdx.x * blockDim.x + threadIdx.x;
    if (t >= nEdges * 16) return;
    int e = t >> 4, k = t & 15;
    int d = dst[e];
    atomicAdd(&aggEA[d * 16 + k], ea[t]);
    if (k == 0) atomicAdd(&deg[d], 1);
}

// single-block chunked exclusive scan of deg -> rowptr (rowptr[n]=total).
__global__ void scan_kernel(const int* __restrict__ deg, int* __restrict__ rowptr, int n) {
    __shared__ int sm[1024];
    int carry = 0;
    for (int base = 0; base < n; base += 1024) {
        int i = base + (int)threadIdx.x;
        int v = (i < n) ? deg[i] : 0;
        sm[threadIdx.x] = v;
        __syncthreads();
        for (int off = 1; off < 1024; off <<= 1) {
            int t = (threadIdx.x >= (unsigned)off) ? sm[threadIdx.x - off] : 0;
            __syncthreads();
            sm[threadIdx.x] += t;
            __syncthreads();
        }
        int incl = sm[threadIdx.x];
        if (i < n) rowptr[i] = carry + incl - v;   // exclusive
        carry += sm[1023];
        __syncthreads();
    }
    if (threadIdx.x == 0) rowptr[n] = carry;
}

__global__ void fill_csr_kernel(const int* __restrict__ src, const int* __restrict__ dst,
                                const int* __restrict__ rowptr, int* __restrict__ fill,
                                int* __restrict__ csr, int nEdges) {
    int e = blockIdx.x * blockDim.x + threadIdx.x;
    if (e >= nEdges) return;
    int d = dst[e];
    int p = atomicAdd(&fill[d], 1);
    csr[rowptr[d] + p] = src[e];
}

// ---------------- per-layer aggregation ----------------
// one wave32 per node; lane owns 4 contiguous features of 128.
__global__ void agg_kernel(const _Float16* __restrict__ hf,
                           const int* __restrict__ rowptr,
                           const int* __restrict__ csr,
                           const float* __restrict__ aggEA,
                           const int* __restrict__ deg,
                           const float* __restrict__ ew,   // [128,16]
                           const float* __restrict__ eb,   // [128]
                           _Float16* __restrict__ aggh,    // [N,256]
                           int nNodes) {
    int gwave = (blockIdx.x * blockDim.x + threadIdx.x) >> 5;
    int lane  = threadIdx.x & 31;
    if (gwave >= nNodes) return;
    int node = gwave;

    // ---- node-feature aggregation (self loop + in-neighbors) ----
    float acc[4];
    {
        v4h s = *(const v4h*)(hf + (size_t)node * 128 + lane * 4);
        #pragma unroll
        for (int i = 0; i < 4; ++i) acc[i] = (float)s[i];
    }
    int beg = rowptr[node], end = rowptr[node + 1];
    for (int e = beg; e < end; ++e) {
        int sn = csr[e];
        if (e + 1 < end)
            __builtin_prefetch(hf + (size_t)csr[e + 1] * 128, 0, 0); // global_prefetch_b8
        v4h t = *(const v4h*)(hf + (size_t)sn * 128 + lane * 4);
        #pragma unroll
        for (int i = 0; i < 4; ++i) acc[i] += (float)t[i];
    }
    {
        v4h o;
        #pragma unroll
        for (int i = 0; i < 4; ++i) o[i] = (_Float16)acc[i];
        *(v4h*)(aggh + (size_t)node * 256 + lane * 4) = o;
    }

    // ---- edge-feature aggregate: aggEA[node] @ ew^T + (deg+1)*eb ----
    float a16[16];
    const float* ar = aggEA + (size_t)node * 16;
    #pragma unroll
    for (int k = 0; k < 16; ++k) a16[k] = ar[k];
    float dp1 = (float)deg[node] + 1.0f;

    v4h o2;
    #pragma unroll
    for (int i = 0; i < 4; ++i) {
        int c = lane * 4 + i;
        float v = dp1 * eb[c];
        const float* wr = ew + (size_t)c * 16;
        #pragma unroll
        for (int k = 0; k < 16; ++k) v += a16[k] * wr[k];
        o2[i] = (_Float16)v;
    }
    *(v4h*)(aggh + (size_t)node * 256 + 128 + lane * 4) = o2;
}

// ---------------- WMMA GEMM 1:  [N,256] x [256,256]^T + b -> f32 ----------------
// block = 128 threads = 4 waves; wave w owns 64 output columns; M-tile = 16 rows.
__global__ void gemm1_kernel(const _Float16* __restrict__ A,   // aggh [N,256] f16
                             const _Float16* __restrict__ Bw,  // w1h  [256,256] f16 (out x in)
                             const float* __restrict__ bias,   // [256]
                             float* __restrict__ Cout) {       // hbig [N,256] f32
    int wave = threadIdx.x >> 5;
    int lane = threadIdx.x & 31;
    int hi = lane >> 4, lm = lane & 15;
    int m0 = blockIdx.x * 16;
    int n0 = wave * 64;

    v8f acc[4] = {};
    #pragma unroll
    for (int kb = 0; kb < 256; kb += 32) {
        FragH a;  // A 16x32 f16 fragment: halves 0-7 K=kb+8*hi.., 8-15 K=kb+16+8*hi..
        const _Float16* ap = A + (size_t)(m0 + lm) * 256 + kb + hi * 8;
        a.u[0] = *(const v4u*)(ap);
        a.u[1] = *(const v4u*)(ap + 16);
        #pragma unroll
        for (int t = 0; t < 4; ++t) {
            FragH b;  // B 32x16: lane holds column n0+t*16+lm, halves 0-7 K=kb+8*hi.., 8-15 K=kb+16+8*hi..
            const _Float16* bp = Bw + (size_t)(n0 + t * 16 + lm) * 256 + kb + hi * 8;
            b.u[0] = *(const v4u*)(bp);
            b.u[1] = *(const v4u*)(bp + 16);
            acc[t] = __builtin_amdgcn_wmma_f32_16x16x32_f16(
                false, a.h, false, b.h, (short)0, acc[t], false, false);
        }
    }
    #pragma unroll
    for (int t = 0; t < 4; ++t) {
        int col = n0 + t * 16 + lm;
        float bb = bias[col];
        #pragma unroll
        for (int r = 0; r < 8; ++r) {
            int row = m0 + r + hi * 8;
            Cout[(size_t)row * 256 + col] = acc[t][r] + bb;
        }
    }
}

// ---------------- batch-norm ----------------
__global__ void bn_stats_kernel(const float* __restrict__ hbig, float* __restrict__ sums,
                                int nRows, int rowsPerBlock) {
    int c = threadIdx.x;                     // 256 columns
    int r0 = blockIdx.x * rowsPerBlock;
    int r1 = r0 + rowsPerBlock; if (r1 > nRows) r1 = nRows;
    float s = 0.0f, q = 0.0f;
    for (int r = r0; r < r1; ++r) {
        float v = hbig[(size_t)r * 256 + c];
        s += v; q += v * v;
    }
    atomicAdd(&sums[c], s);
    atomicAdd(&sums[256 + c], q);
}

__global__ void bn_finalize_kernel(const float* __restrict__ sums,
                                   const float* __restrict__ bnw,
                                   const float* __restrict__ bnb,
                                   float* __restrict__ scale, float* __restrict__ shift,
                                   float invN) {
    int c = threadIdx.x;
    float mean = sums[c] * invN;
    float var  = sums[256 + c] * invN - mean * mean;
    float rstd = rsqrtf(var + 1e-5f);
    float sc = bnw[c] * rstd;
    scale[c] = sc;
    shift[c] = bnb[c] - mean * sc;
}

__global__ void bn_apply_kernel(const float* __restrict__ hbig,
                                const float* __restrict__ scale,
                                const float* __restrict__ shift,
                                _Float16* __restrict__ hact, int total) {
    int i = blockIdx.x * blockDim.x + threadIdx.x;
    if (i >= total) return;
    int c = i & 255;
    float v = fmaf(hbig[i], scale[c], shift[c]);
    hact[i] = (_Float16)fmaxf(v, 0.0f);
}

// ---------------- WMMA GEMM 2:  [N,256] x [128,256]^T + b ----------------
// block = 128 threads = 4 waves; wave w owns 32 output columns.
__global__ void gemm2_kernel(const _Float16* __restrict__ A,   // hact [N,256] f16
                             const _Float16* __restrict__ Bw,  // w2h [128,256] f16
                             const float* __restrict__ bias,   // [128]
                             _Float16* __restrict__ outH,      // hf [N,128] (layers 0-3)
                             float* __restrict__ outF,         // d_out (layer 4)
                             int lastLayer) {
    int wave = threadIdx.x >> 5;
    int lane = threadIdx.x & 31;
    int hi = lane >> 4, lm = lane & 15;
    int m0 = blockIdx.x * 16;
    int n0 = wave * 32;

    v8f acc[2] = {};
    #pragma unroll
    for (int kb = 0; kb < 256; kb += 32) {
        FragH a;
        const _Float16* ap = A + (size_t)(m0 + lm) * 256 + kb + hi * 8;
        a.u[0] = *(const v4u*)(ap);
        a.u[1] = *(const v4u*)(ap + 16);
        #pragma unroll
        for (int t = 0; t < 2; ++t) {
            FragH b;
            const _Float16* bp = Bw + (size_t)(n0 + t * 16 + lm) * 256 + kb + hi * 8;
            b.u[0] = *(const v4u*)(bp);
            b.u[1] = *(const v4u*)(bp + 16);
            acc[t] = __builtin_amdgcn_wmma_f32_16x16x32_f16(
                false, a.h, false, b.h, (short)0, acc[t], false, false);
        }
    }
    #pragma unroll
    for (int t = 0; t < 2; ++t) {
        int col = n0 + t * 16 + lm;
        float bb = bias[col];
        #pragma unroll
        for (int r = 0; r < 8; ++r) {
            int row = m0 + r + hi * 8;
            float v = acc[t][r] + bb;
            if (lastLayer) {
                outF[(size_t)row * 128 + col] = v;
            } else {
                outH[(size_t)row * 128 + col] = (_Float16)fmaxf(v, 0.0f);
            }
        }
    }
}

// ---------------------------------------------------------------------------
extern "C" void kernel_launch(void* const* d_in, const int* in_sizes, int n_in,
                              void* d_out, int out_size, void* d_ws, size_t ws_size,
                              hipStream_t stream) {
    (void)in_sizes; (void)n_in; (void)out_size; (void)ws_size;

    const float* x         = (const float*)d_in[0];
    const float* edge_attr = (const float*)d_in[1];
    const float* W1        = (const float*)d_in[2];   // [5,256,256]
    const float* B1        = (const float*)d_in[3];   // [5,256]
    const float* BNW       = (const float*)d_in[4];   // [5,256]
    const float* BNB       = (const float*)d_in[5];   // [5,256]
    const float* W2        = (const float*)d_in[6];   // [5,128,256]
    const float* B2        = (const float*)d_in[7];   // [5,128]
    const float* EW        = (const float*)d_in[8];   // [5,128,16]
    const float* EB        = (const float*)d_in[9];   // [5,128]
    const int*   eidx      = (const int*)d_in[10];    // [2,E]

    const int N = N_NODES, E = N_EDGES;
    const int* srcI = eidx;
    const int* dstI = eidx + E;

    char* ws = (char*)d_ws;
    size_t off = 0;
    auto alloc = [&](size_t bytes) -> char* {
        char* p = ws + off;
        off += (bytes + 255) & ~(size_t)255;
        return p;
    };
    _Float16* hf    = (_Float16*)alloc((size_t)N * 128 * 2);   // node features f16
    _Float16* aggh  = (_Float16*)alloc((size_t)N * 256 * 2);   // concat(agg_x, agg_e) f16
    float*    hbig  = (float*)   alloc((size_t)N * 256 * 4);   // pre-BN activations
    _Float16* hact  = (_Float16*)alloc((size_t)N * 256 * 2);   // post BN+relu f16
    float*    aggEA = (float*)   alloc((size_t)N * 16 * 4);
    int*      deg   = (int*)     alloc((size_t)N * 4);
    int*      rowp  = (int*)     alloc((size_t)(N + 1) * 4);
    int*      fill  = (int*)     alloc((size_t)N * 4);
    int*      csr   = (int*)     alloc((size_t)E * 4);
    _Float16* w1h   = (_Float16*)alloc((size_t)256 * 256 * 2);
    _Float16* w2h   = (_Float16*)alloc((size_t)128 * 256 * 2);
    float*    sums  = (float*)   alloc((size_t)512 * 4);
    float*    scale = (float*)   alloc((size_t)256 * 4);
    float*    shift = (float*)   alloc((size_t)256 * 4);

    const int TB = 256;
    // ---- one-time graph preprocessing ----
    zero_f32_kernel<<<(N * 16 + TB - 1) / TB, TB, 0, stream>>>(aggEA, N * 16);
    zero_i32_kernel<<<(N + TB - 1) / TB, TB, 0, stream>>>(deg, N);
    zero_i32_kernel<<<(N + TB - 1) / TB, TB, 0, stream>>>(fill, N);
    edge_accum_kernel<<<(E * 16 + TB - 1) / TB, TB, 0, stream>>>(dstI, edge_attr, aggEA, deg, E);
    scan_kernel<<<1, 1024, 0, stream>>>(deg, rowp, N);
    fill_csr_kernel<<<(E + TB - 1) / TB, TB, 0, stream>>>(srcI, dstI, rowp, fill, csr, E);
    cvt_f32_to_f16_kernel<<<(N * 128 + TB - 1) / TB, TB, 0, stream>>>(x, hf, N * 128);

    // ---- 5 GIN layers ----
    for (int l = 0; l < 5; ++l) {
        cvt_f32_to_f16_kernel<<<(256 * 256 + TB - 1) / TB, TB, 0, stream>>>(
            W1 + (size_t)l * 256 * 256, w1h, 256 * 256);
        cvt_f32_to_f16_kernel<<<(128 * 256 + TB - 1) / TB, TB, 0, stream>>>(
            W2 + (size_t)l * 128 * 256, w2h, 128 * 256);

        // aggregation: 8 waves (nodes) per 256-thread block
        agg_kernel<<<N / 8, 256, 0, stream>>>(hf, rowp, csr, aggEA, deg,
                                              EW + (size_t)l * 128 * 16,
                                              EB + (size_t)l * 128, aggh, N);

        gemm1_kernel<<<N / 16, 128, 0, stream>>>(aggh, w1h, B1 + (size_t)l * 256, hbig);

        zero_f32_kernel<<<2, 256, 0, stream>>>(sums, 512);
        bn_stats_kernel<<<400, 256, 0, stream>>>(hbig, sums, N, 250);
        bn_finalize_kernel<<<1, 256, 0, stream>>>(sums, BNW + (size_t)l * 256,
                                                  BNB + (size_t)l * 256,
                                                  scale, shift, 1.0f / (float)N);
        bn_apply_kernel<<<(N * 256 + TB - 1) / TB, TB, 0, stream>>>(hbig, scale, shift,
                                                                    hact, N * 256);

        gemm2_kernel<<<N / 16, 128, 0, stream>>>(hact, w2h, B2 + (size_t)l * 128,
                                                 hf, (float*)d_out, (l == 4) ? 1 : 0);
    }
}
